// Scout_68015102100198
// MI455X (gfx1250) — matrix operand order repaired and checked by reading, not compiled, not run on hardware
//
#include <hip/hip_runtime.h>
#include <hip/hip_bf16.h>
#include <math.h>

typedef _Float16 h16;
typedef __attribute__((ext_vector_type(16))) _Float16 v16h;
typedef __attribute__((ext_vector_type(8)))  _Float16 v8h;
typedef __attribute__((ext_vector_type(8)))  float    v8f;

#define DMODEL  768
#define NHEAD   12
#define NLAYER  6
#define HEADDIM 64
#define DFF     3072
#define BB      4
#define NN      512
#define M2      (BB*NN)      /* 2048 tokens */
#define RDIM    384
#define THEADS  72           /* NLAYER*NHEAD */

#define LDSROW  40           /* padded B-tile row stride in halves (80B):   */
                             /* 80B = 20 banks -> 16 rows span 16 distinct banks */

// ---------------------------------------------------------------------------
// CDNA5 async copy global->LDS (no VGPR round trip, tracked by ASYNCcnt).
// LDS byte address = low 32 bits of the generic pointer (LDS aperture maps
// addr[31:0] to the LDS offset).
// ---------------------------------------------------------------------------
__device__ __forceinline__ void async_copy_b128(void* lds_dst, const void* gsrc) {
  asm volatile("global_load_async_to_lds_b128 %0, %1, off"
               :: "v"((unsigned)(uintptr_t)lds_dst), "v"(gsrc)
               : "memory");
}
__device__ __forceinline__ void wait_async() {
  asm volatile("s_wait_asynccnt 0x0" ::: "memory");
}

// ---------------------------------------------------------------------------
// Fragment builders (CDNA5 16-bit WMMA VGPR layouts, wave32)
// A 16x32: lanes 0-15 hold M=lane, K in {0..7}U{16..23}; lanes 16-31 K in {8..15}U{24..31}
// B 32x16: lanes 0-15 hold N=lane, K in {0..7}U{16..23}; lanes 16-31 K in {8..15}U{24..31}
// ---------------------------------------------------------------------------
__device__ __forceinline__ v16h cat16(v8h lo, v8h hi) {
  return __builtin_shufflevector(lo, hi, 0,1,2,3,4,5,6,7,8,9,10,11,12,13,14,15);
}
__device__ __forceinline__ v16h load_frag_a(const h16* rowp, int half) {
  const h16* p = rowp + half * 8;
  return cat16(*(const v8h*)(p), *(const v8h*)(p + 16));
}

// ---------------------------------------------------------------------------
// Batched WMMA GEMM:  C[z] = A[z] (MxK f16) * W[z]^T (N rows of K, f16)
//                            + bias[n] [+ resid[m,n]]
// Wave tile 32x64 (8 accums, each B fragment reused twice); block 8 waves =
// 256x64.  B tile (64x32) streamed into double-buffered LDS with async-to-LDS
// copies; A fragments register-prefetched one k-step ahead.
// ---------------------------------------------------------------------------
__global__ __launch_bounds__(256) void wmma_gemm(
    const h16* __restrict__ A, size_t batchA, int lda,
    const h16* __restrict__ W, size_t batchW, int ldw,
    float* __restrict__ C, size_t batchC, int ldc,
    const float* __restrict__ bias,
    const float* __restrict__ resid,
    int M, int N, int K)
{
  __shared__ h16 smB[2][64 * LDSROW];   // 2 x 5120 B

  const int z = blockIdx.z;
  A += batchA * (size_t)z;
  W += batchW * (size_t)z;
  C += batchC * (size_t)z;
  if (resid) resid += batchC * (size_t)z;

  const int tid  = threadIdx.x;
  const int wave = tid >> 5;
  const int lane = tid & 31;
  const int half = lane >> 4;
  const int lh   = lane & 15;
  const int m0 = blockIdx.x * 256 + wave * 32;
  const int n0 = blockIdx.y * 64;

  // cooperative B staging: 64 rows x 4 chunks of 8 halves = 256 chunks
  const int brow   = tid >> 2;            // 0..63
  const int bchunk = tid & 3;             // 0..3
  int wr = n0 + brow; if (wr >= N) wr = 0;          // clamped; store is guarded
  const h16* wsrc = W + (size_t)wr * ldw + bchunk * 8;
  h16* wdst0 = &smB[0][brow * LDSROW + bchunk * 8];
  h16* wdst1 = &smB[1][brow * LDSROW + bchunk * 8];

  int arow0 = m0 + lh;      if (arow0 > M - 1) arow0 = M - 1;
  int arow1 = m0 + 16 + lh; if (arow1 > M - 1) arow1 = M - 1;
  const h16* ap0 = A + (size_t)arow0 * lda;
  const h16* ap1 = A + (size_t)arow1 * lda;

  v8f acc[2][4] = {};

  // prologue: async-stage B tile k=0, A fragments k=0 in registers
  async_copy_b128(wdst0, wsrc);
  v8h alo0 = *(const v8h*)(ap0 + half * 8);
  v8h ahi0 = *(const v8h*)(ap0 + half * 8 + 16);
  v8h alo1 = *(const v8h*)(ap1 + half * 8);
  v8h ahi1 = *(const v8h*)(ap1 + half * 8 + 16);

  for (int k0 = 0; k0 < K; k0 += 32) {
    const h16* smb = (k0 & 32) ? smB[1] : smB[0];

    wait_async();              // own chunk landed in LDS
    __syncthreads();           // whole tile valid

    const int k1 = k0 + 32;
    if (k1 < K) {              // background-stage next B tile into other buffer
      async_copy_b128((k1 & 32) ? wdst1 : wdst0, wsrc + k1);
      __builtin_prefetch(wsrc + k1 + 32, 0, 3);   // pull k+2 line toward L2
    }

    v16h af0 = cat16(alo0, ahi0);
    v16h af1 = cat16(alo1, ahi1);
    if (k1 < K) {              // register-prefetch next A fragments
      alo0 = *(const v8h*)(ap0 + k1 + half * 8);
      ahi0 = *(const v8h*)(ap0 + k1 + half * 8 + 16);
      alo1 = *(const v8h*)(ap1 + k1 + half * 8);
      ahi1 = *(const v8h*)(ap1 + k1 + half * 8 + 16);
    }

#pragma unroll
    for (int t = 0; t < 4; ++t) {
      const h16* p = smb + (size_t)(t * 16 + lh) * LDSROW + half * 8;
      v16h bf = cat16(*(const v8h*)(p), *(const v8h*)(p + 16));  // ds_load_b128 x2
      acc[0][t] = __builtin_amdgcn_wmma_f32_16x16x32_f16(
          false, af0, false, bf, (short)0, acc[0][t], false, false);
      acc[1][t] = __builtin_amdgcn_wmma_f32_16x16x32_f16(
          false, af1, false, bf, (short)0, acc[1][t], false, false);
    }
  }

#pragma unroll
  for (int g = 0; g < 2; ++g) {
#pragma unroll
    for (int t = 0; t < 4; ++t) {
      const int col = n0 + t * 16 + lh;
      if (col >= N) continue;
      const float badd = bias ? bias[col] : 0.0f;
#pragma unroll
      for (int r = 0; r < 8; ++r) {
        const int row = m0 + g * 16 + half * 8 + r;
        if (row < M) {
          const size_t idx = (size_t)row * ldc + col;
          float v = acc[g][t][r] + badd;
          if (resid) v += resid[idx];
          C[idx] = v;
        }
      }
    }
  }
}

// ---------------------------------------------------------------------------
// Elementwise / row kernels
// ---------------------------------------------------------------------------
__global__ void f32_to_f16_kernel(const float* __restrict__ s, h16* __restrict__ d, size_t n)
{
  size_t i = (size_t)blockIdx.x * blockDim.x + threadIdx.x;
  const size_t stride = (size_t)gridDim.x * blockDim.x;
  for (; i < n; i += stride) d[i] = (h16)s[i];
}

__global__ void gelu_f16_kernel(const float* __restrict__ s, h16* __restrict__ d, size_t n)
{
  size_t i = (size_t)blockIdx.x * blockDim.x + threadIdx.x;
  const size_t stride = (size_t)gridDim.x * blockDim.x;
  for (; i < n; i += stride) {
    float xv = s[i];
    d[i] = (h16)(0.5f * xv * (1.0f + erff(xv * 0.70710678118654752f)));
  }
}

// LayerNorm over D=768, one block (256 thr) per token row; writes f16.
__global__ __launch_bounds__(256) void ln_kernel(
    const float* __restrict__ x, const float* __restrict__ g,
    const float* __restrict__ be, h16* __restrict__ out)
{
  __shared__ float red[256];
  const float* xr = x + (size_t)blockIdx.x * DMODEL;
  h16* orow = out + (size_t)blockIdx.x * DMODEL;

  float s = 0.f;
  for (int i = threadIdx.x; i < DMODEL; i += 256) s += xr[i];
  red[threadIdx.x] = s; __syncthreads();
  for (int off = 128; off > 0; off >>= 1) {
    if (threadIdx.x < off) red[threadIdx.x] += red[threadIdx.x + off];
    __syncthreads();
  }
  const float mu = red[0] * (1.0f / DMODEL);
  __syncthreads();

  float v = 0.f;
  for (int i = threadIdx.x; i < DMODEL; i += 256) { float dd = xr[i] - mu; v += dd * dd; }
  red[threadIdx.x] = v; __syncthreads();
  for (int off = 128; off > 0; off >>= 1) {
    if (threadIdx.x < off) red[threadIdx.x] += red[threadIdx.x + off];
    __syncthreads();
  }
  const float rstd = rsqrtf(red[0] * (1.0f / DMODEL) + 1e-5f);

  for (int i = threadIdx.x; i < DMODEL; i += 256)
    orow[i] = (h16)((xr[i] - mu) * rstd * g[i] + be[i]);
}

// sigmoid(scale*scores) -> f16 probs, per-row sum (+1e-6) -> rsum.
__global__ __launch_bounds__(256) void sigmoid_rows_kernel(
    const float* __restrict__ S, h16* __restrict__ P, float* __restrict__ rs)
{
  __shared__ float red[256];
  const float* srow = S + (size_t)blockIdx.x * NN;
  h16* prow = P + (size_t)blockIdx.x * NN;
  float s = 0.f;
  for (int i = threadIdx.x; i < NN; i += 256) {
    float p = 1.0f / (1.0f + __expf(-srow[i] * 0.125f));  // 1/sqrt(64)
    prow[i] = (h16)p;
    s += p;
  }
  red[threadIdx.x] = s; __syncthreads();
  for (int off = 128; off > 0; off >>= 1) {
    if (threadIdx.x < off) red[threadIdx.x] += red[threadIdx.x + off];
    __syncthreads();
  }
  if (threadIdx.x == 0) rs[blockIdx.x] = red[0] + 1e-6f;
}

// [B,N,H,D] f32 -> [B,H,N,D] f16
__global__ void split_heads_kernel(const float* __restrict__ src, h16* __restrict__ dst)
{
  size_t i = (size_t)blockIdx.x * blockDim.x + threadIdx.x;
  if (i >= (size_t)M2 * DMODEL) return;
  const int d = (int)(i & 63);
  size_t t = i >> 6;
  const int h = (int)(t % NHEAD); t /= NHEAD;
  const int n = (int)(t % NN);
  const int b = (int)(t / NN);
  dst[((((size_t)b * NHEAD + h) * NN + n) << 6) + d] = (h16)src[i];
}

// [B,N,H,D] f32 -> [B,H,D,N] f16  (V transposed for PV gemm)
__global__ void split_heads_T_kernel(const float* __restrict__ src, h16* __restrict__ dst)
{
  size_t i = (size_t)blockIdx.x * blockDim.x + threadIdx.x;
  if (i >= (size_t)M2 * DMODEL) return;
  const int d = (int)(i & 63);
  size_t t = i >> 6;
  const int h = (int)(t % NHEAD); t /= NHEAD;
  const int n = (int)(t % NN);
  const int b = (int)(t / NN);
  dst[(((size_t)b * NHEAD + h) * HEADDIM + d) * NN + n] = (h16)src[i];
}

// attn [B,H,N,D] f32 / rowsum -> [B,N,H*D] f16
__global__ void attn_fix_kernel(const float* __restrict__ src,
                                const float* __restrict__ rs, h16* __restrict__ dst)
{
  size_t i = (size_t)blockIdx.x * blockDim.x + threadIdx.x;
  if (i >= (size_t)M2 * DMODEL) return;
  const int d = (int)(i & 63);
  size_t t = i >> 6;                 // t == (b*H + h)*N + n  == rowsum index
  const float v = src[i] / rs[t];
  const int n = (int)(t % NN); t /= NN;
  const int h = (int)(t % NHEAD);
  const int b = (int)(t / NHEAD);
  dst[((size_t)b * NN + n) * DMODEL + h * HEADDIM + d] = (h16)v;
}

// ---------------------------------------------------------------------------
// Final aggregation: out[b,n,m] = sum_{j=0..71} raw[b,n,j] *
//      sigmoid(scale * q_j[b,n,:].k_j[b,m,:])  (diagonal zeroed) + agg_bias.
// WMMA-recomputes all 72 score tiles. K-tile (64x64 f16, shared by all 8
// waves of the block) streamed via async-to-LDS double buffer, one head ahead.
// qh,kh: [L,B,H,N,64] f16.  Wave tile 16x64, block 128x64, grid (4,8,B).
// ---------------------------------------------------------------------------
#define KROW 72   /* padded K-tile row stride (144B) */

__global__ __launch_bounds__(256) void final_agg_kernel(
    const h16* __restrict__ qh, const h16* __restrict__ kh,
    const float* __restrict__ raw, const float* __restrict__ agg_bias,
    float* __restrict__ out)
{
  __shared__ float wsm[128 * THEADS];     // 36 KB raw_w for this block's rows
  __shared__ h16   smK[2][64 * KROW];     // 2 x 9 KB K-tile double buffer

  const int b = blockIdx.z;
  const int tid  = threadIdx.x;
  const int wave = tid >> 5;
  const int lane = tid & 31;
  const int half = lane >> 4;
  const int lh   = lane & 15;
  const int n0 = blockIdx.x * 128 + wave * 16;
  const int m0 = blockIdx.y * 64;

  // cooperative K staging: 64 rows x 8 chunks of 8 halves = 512 chunks, 2/thread
  const int r0 = tid >> 3,        o0 = (tid & 7) * 8;
  const int r1 = (tid >> 3) + 32, o1 = o0;
  h16* sd[2][2] = { { &smK[0][r0 * KROW + o0], &smK[0][r1 * KROW + o1] },
                    { &smK[1][r0 * KROW + o0], &smK[1][r1 * KROW + o1] } };

  // prologue: async-stage head 0's K tile
  const h16* kb0 = kh + ((size_t)b * NHEAD) * NN * HEADDIM;   // l=0,h=0
  async_copy_b128(sd[0][0], kb0 + (size_t)(m0 + r0) * HEADDIM + o0);
  async_copy_b128(sd[0][1], kb0 + (size_t)(m0 + r1) * HEADDIM + o1);

  for (int i = tid; i < 128 * THEADS; i += 256) {
    const int rrow = blockIdx.x * 128 + i / THEADS;
    wsm[i] = raw[((size_t)b * NN + rrow) * THEADS + (i % THEADS)];
  }

  float acc[4][8] = {};
  const int qrow = n0 + lh;

  for (int j = 0; j < THEADS; ++j) {
    const h16* sk = smK[j & 1];
    wait_async();
    __syncthreads();

    if (j + 1 < THEADS) {                 // background-stage next head's K tile
      const int l1 = (j + 1) / NHEAD, h1 = (j + 1) % NHEAD;
      const h16* kn = kh + (((size_t)l1 * BB + b) * NHEAD + h1) * NN * HEADDIM;
      async_copy_b128(sd[(j + 1) & 1][0], kn + (size_t)(m0 + r0) * HEADDIM + o0);
      async_copy_b128(sd[(j + 1) & 1][1], kn + (size_t)(m0 + r1) * HEADDIM + o1);
    }

    const int l = j / NHEAD, h = j % NHEAD;
    const h16* qb = qh + (((size_t)l * BB + b) * NHEAD + h) * NN * HEADDIM;

    v8f s[4] = {};
#pragma unroll
    for (int k0 = 0; k0 < HEADDIM; k0 += 32) {
      v16h af = load_frag_a(qb + (size_t)qrow * HEADDIM + k0, half);
      v16h bf[4];
#pragma unroll
      for (int t = 0; t < 4; ++t) {
        const h16* p = sk + (size_t)(t * 16 + lh) * KROW + k0 + half * 8;
        bf[t] = cat16(*(const v8h*)(p), *(const v8h*)(p + 16));
      }
#pragma unroll
      for (int t = 0; t < 4; ++t)
        s[t] = __builtin_amdgcn_wmma_f32_16x16x32_f16(
            false, af, false, bf[t], (short)0, s[t], false, false);
    }
#pragma unroll
    for (int r = 0; r < 8; ++r) {
      const float w = wsm[(wave * 16 + half * 8 + r) * THEADS + j];
#pragma unroll
      for (int t = 0; t < 4; ++t)
        acc[t][r] += w / (1.0f + __expf(-s[t][r] * 0.125f));
    }
  }

  const float ab = agg_bias[0];
#pragma unroll
  for (int t = 0; t < 4; ++t) {
    const int m = m0 + t * 16 + lh;
#pragma unroll
    for (int r = 0; r < 8; ++r) {
      const int n = n0 + half * 8 + r;
      const float v = (n == m) ? 0.0f : acc[t][r];
      out[((size_t)b * NN + n) * NN + m] = v + ab;
    }
  }
}

// ---------------------------------------------------------------------------
// Host side
// ---------------------------------------------------------------------------
extern "C" void kernel_launch(void* const* d_in, const int* in_sizes, int n_in,
                              void* d_out, int out_size, void* d_ws, size_t ws_size,
                              hipStream_t stream)
{
  (void)in_sizes; (void)n_in; (void)out_size; (void)ws_size;

  const float* emb  = (const float*)d_in[0];
  const float* Win  = (const float*)d_in[1];
  const float* b_in = (const float*)d_in[2];
  const float* Wq   = (const float*)d_in[3];
  const float* bq   = (const float*)d_in[4];
  const float* Wk   = (const float*)d_in[5];
  const float* bk   = (const float*)d_in[6];
  const float* Wv   = (const float*)d_in[7];
  const float* bv   = (const float*)d_in[8];
  const float* Wo   = (const float*)d_in[9];
  const float* bo   = (const float*)d_in[10];
  const float* g1   = (const float*)d_in[11];
  const float* be1  = (const float*)d_in[12];
  const float* g2   = (const float*)d_in[13];
  const float* be2  = (const float*)d_in[14];
  const float* W1   = (const float*)d_in[15];
  const float* bf1  = (const float*)d_in[16];
  const float* W2   = (const float*)d_in[17];
  const float* bf2  = (const float*)d_in[18];
  const float* Wr1  = (const float*)d_in[19];
  const float* br1  = (const float*)d_in[20];
  const float* Wr2  = (const float*)d_in[21];
  const float* br2  = (const float*)d_in[22];
  const float* aggb = (const float*)d_in[23];
  float* out = (float*)d_out;

  // ---- workspace carve-out (256B aligned) ----
  char* wsp = (char*)d_ws;
  size_t off = 0;
  auto alloc_h = [&](size_t elems) -> h16* {
    h16* p = (h16*)(wsp + off); off += ((elems * 2 + 255) & ~(size_t)255); return p;
  };
  auto alloc_f = [&](size_t elems) -> float* {
    float* p = (float*)(wsp + off); off += ((elems * 4 + 255) & ~(size_t)255); return p;
  };

  const size_t DD   = (size_t)DMODEL * DMODEL;
  const size_t QKSZ = (size_t)BB * NHEAD * NN * HEADDIM;   // per-layer q/k f16

  h16* emb_h = alloc_h((size_t)M2 * DMODEL);
  h16* Win_h = alloc_h(DD);
  h16* Wq_h  = alloc_h((size_t)NLAYER * DD);
  h16* Wk_h  = alloc_h((size_t)NLAYER * DD);
  h16* Wv_h  = alloc_h((size_t)NLAYER * DD);
  h16* Wo_h  = alloc_h((size_t)NLAYER * DD);
  h16* W1_h  = alloc_h((size_t)NLAYER * DFF * DMODEL);
  h16* W2_h  = alloc_h((size_t)NLAYER * DMODEL * DFF);
  h16* Wr1_h = alloc_h((size_t)RDIM * DMODEL);
  h16* Wr2_h = alloc_h((size_t)THEADS * RDIM);
  h16* n1_h  = alloc_h((size_t)M2 * DMODEL);
  h16* n2_h  = alloc_h((size_t)M2 * DMODEL);
  h16* qh    = alloc_h((size_t)NLAYER * QKSZ);
  h16* kh    = alloc_h((size_t)NLAYER * QKSZ);
  h16* vT    = alloc_h(QKSZ);
  h16* probs = alloc_h((size_t)BB * NHEAD * NN * NN);
  h16* attn_h= alloc_h((size_t)M2 * DMODEL);
  h16* ffg_h = alloc_h((size_t)M2 * DFF);
  h16* x_h   = alloc_h((size_t)M2 * DMODEL);
  h16* r1g_h = alloc_h((size_t)M2 * RDIM);
  float* x    = alloc_f((size_t)M2 * DMODEL);
  float* tf   = alloc_f((size_t)BB * NHEAD * NN * NN);   // big reusable f32 scratch
  float* rsum = alloc_f((size_t)BB * NHEAD * NN);

  auto conv = [&](const float* s, h16* d, size_t n) {
    int blocks = (int)((n + 1023) / 1024); if (blocks > 4096) blocks = 4096;
    f32_to_f16_kernel<<<blocks, 256, 0, stream>>>(s, d, n);
  };
  auto gelu = [&](const float* s, h16* d, size_t n) {
    int blocks = (int)((n + 1023) / 1024); if (blocks > 4096) blocks = 4096;
    gelu_f16_kernel<<<blocks, 256, 0, stream>>>(s, d, n);
  };
  auto gemm = [&](const h16* A, size_t bA, int lda, const h16* Wm, size_t bW, int ldw,
                  float* Cm, size_t bC, int ldc, const float* bias, const float* resid,
                  int M, int N, int K, int batches) {
    dim3 grid((M + 255) / 256, (N + 63) / 64, batches);
    wmma_gemm<<<grid, 256, 0, stream>>>(A, bA, lda, Wm, bW, ldw, Cm, bC, ldc,
                                        bias, resid, M, N, K);
  };

  // ---- weight / input conversion to f16 (weights stay L2-resident: ~85MB) ----
  conv(emb, emb_h, (size_t)M2 * DMODEL);
  conv(Win, Win_h, DD);
  conv(Wq, Wq_h, (size_t)NLAYER * DD);
  conv(Wk, Wk_h, (size_t)NLAYER * DD);
  conv(Wv, Wv_h, (size_t)NLAYER * DD);
  conv(Wo, Wo_h, (size_t)NLAYER * DD);
  conv(W1, W1_h, (size_t)NLAYER * DFF * DMODEL);
  conv(W2, W2_h, (size_t)NLAYER * DMODEL * DFF);
  conv(Wr1, Wr1_h, (size_t)RDIM * DMODEL);
  conv(Wr2, Wr2_h, (size_t)THEADS * RDIM);

  // ---- input projection: x = emb @ Win^T + b_in ----
  gemm(emb_h, 0, DMODEL, Win_h, 0, DMODEL, x, 0, DMODEL, b_in, nullptr,
       M2, DMODEL, DMODEL, 1);

  const int ew_blocks = (int)(((size_t)M2 * DMODEL + 255) / 256);  // 6144

  for (int i = 0; i < NLAYER; ++i) {
    const size_t wO = (size_t)i * DD;
    // n1 = LN(x)
    ln_kernel<<<M2, 256, 0, stream>>>(x, g1 + i * DMODEL, be1 + i * DMODEL, n1_h);

    // Q
    gemm(n1_h, 0, DMODEL, Wq_h + wO, 0, DMODEL, tf, 0, DMODEL,
         bq + i * DMODEL, nullptr, M2, DMODEL, DMODEL, 1);
    split_heads_kernel<<<ew_blocks, 256, 0, stream>>>(tf, qh + (size_t)i * QKSZ);
    // K
    gemm(n1_h, 0, DMODEL, Wk_h + wO, 0, DMODEL, tf, 0, DMODEL,
         bk + i * DMODEL, nullptr, M2, DMODEL, DMODEL, 1);
    split_heads_kernel<<<ew_blocks, 256, 0, stream>>>(tf, kh + (size_t)i * QKSZ);
    // V (transposed per head)
    gemm(n1_h, 0, DMODEL, Wv_h + wO, 0, DMODEL, tf, 0, DMODEL,
         bv + i * DMODEL, nullptr, M2, DMODEL, DMODEL, 1);
    split_heads_T_kernel<<<ew_blocks, 256, 0, stream>>>(tf, vT);

    // scores[z] = Q K^T  (48 batches of 512x512x64)
    gemm(qh + (size_t)i * QKSZ, (size_t)NN * HEADDIM, HEADDIM,
         kh + (size_t)i * QKSZ, (size_t)NN * HEADDIM, HEADDIM,
         tf, (size_t)NN * NN, NN, nullptr, nullptr, NN, NN, HEADDIM, BB * NHEAD);

    // probs = sigmoid(scale*scores), rowsums
    sigmoid_rows_kernel<<<BB * NHEAD * NN, 256, 0, stream>>>(tf, probs, rsum);

    // attn = probs @ V   (48 batches of 512x64x512)  -> tf as [B,H,N,D]
    gemm(probs, (size_t)NN * NN, NN, vT, (size_t)HEADDIM * NN, NN,
         tf, (size_t)NN * HEADDIM, HEADDIM, nullptr, nullptr, NN, HEADDIM, NN, BB * NHEAD);

    // divide by rowsums, merge heads -> attn_h [B,N,768] f16
    attn_fix_kernel<<<ew_blocks, 256, 0, stream>>>(tf, rsum, attn_h);

    // x = x + attn @ Wo^T + bo
    gemm(attn_h, 0, DMODEL, Wo_h + wO, 0, DMODEL, x, 0, DMODEL,
         bo + i * DMODEL, x, M2, DMODEL, DMODEL, 1);

    // FFN
    ln_kernel<<<M2, 256, 0, stream>>>(x, g2 + i * DMODEL, be2 + i * DMODEL, n2_h);
    gemm(n2_h, 0, DMODEL, W1_h + (size_t)i * DFF * DMODEL, 0, DMODEL,
         tf, 0, DFF, bf1 + i * DFF, nullptr, M2, DFF, DMODEL, 1);
    gelu(tf, ffg_h, (size_t)M2 * DFF);
    gemm(ffg_h, 0, DFF, W2_h + (size_t)i * DMODEL * DFF, 0, DFF,
         x, 0, DMODEL, bf2 + i * DMODEL, x, M2, DMODEL, DFF, 1);
  }

  // raw_w = gelu(x @ Wr1^T + br1) @ Wr2^T + br2
  conv(x, x_h, (size_t)M2 * DMODEL);
  gemm(x_h, 0, DMODEL, Wr1_h, 0, DMODEL, tf, 0, RDIM, br1, nullptr,
       M2, RDIM, DMODEL, 1);
  gelu(tf, r1g_h, (size_t)M2 * RDIM);
  gemm(r1g_h, 0, RDIM, Wr2_h, 0, RDIM, tf, 0, THEADS, br2, nullptr,
       M2, THEADS, RDIM, 1);

  // final: WMMA-recompute all 72 score tiles, weight, zero diagonal, + bias
  final_agg_kernel<<<dim3(NN / 128, NN / 64, BB), 256, 0, stream>>>(
      qh, kh, tf, aggb, out);
}
